// TFXNetRelativeAttention_82703890252438
// MI455X (gfx1250) — matrix operand order repaired and checked by reading, not compiled, not run on hardware
//
#include <hip/hip_runtime.h>

// ---------------- problem constants ----------------
#define QLEN   1024
#define MLEN   1024
#define KLENC  2048          // QLEN + MLEN
#define RLENC  3072          // KLEN + QLEN
#define BATCH  2
#define NHD    16
#define DHD    64
#define DMODEL 1024
#define SCALEF 0.125f        // 1/sqrt(64)

// ---------------- types ----------------
typedef __attribute__((ext_vector_type(16))) __bf16 v16bf;
typedef __attribute__((ext_vector_type(8)))  __bf16 v8bf;
typedef __attribute__((ext_vector_type(8)))  float  v8f;

// ---------------- helpers ----------------
__device__ __forceinline__ unsigned short f2bf(float f) {
  unsigned int u = __float_as_uint(f);
  unsigned int r = u + 0x7FFFu + ((u >> 16) & 1u);   // RNE
  return (unsigned short)(r >> 16);
}
__device__ __forceinline__ float bf2f(unsigned short h) {
  return __uint_as_float(((unsigned int)h) << 16);
}
__device__ __forceinline__ v8f zero_v8f() {
  v8f z = {0.f, 0.f, 0.f, 0.f, 0.f, 0.f, 0.f, 0.f};
  return z;
}
__device__ __forceinline__ v8f wmma_bf16(v16bf a, v16bf b, v8f c) {
  return __builtin_amdgcn_wmma_f32_16x16x32_bf16(false, a, false, b, (short)0, c, false, false);
}

// A fragment, 16x32 bf16 (contiguous k order, same as B): lane L -> row L%16,
// K contiguous 16 elements at k0+16*(L/16) .. +15.  Two 16B contiguous loads.
__device__ __forceinline__ v16bf load_afrag(const unsigned short* base, int ldk,
                                            int row0, int k0, int lane) {
  const int r    = row0 + (lane & 15);
  const int koff = k0 + ((lane >> 4) << 4);
  const unsigned short* p = base + (size_t)r * ldk + koff;
  v8bf lo = *(const v8bf*)(p);
  v8bf hi = *(const v8bf*)(p + 8);
  v16bf a;
#pragma unroll
  for (int t = 0; t < 8; ++t) { a[t] = lo[t]; a[t + 8] = hi[t]; }
  return a;
}

// B fragment, 32x16 bf16: lane L -> column col0 + L%16,
// K contiguous 16 elements starting at k0 + 16*(L/16).  One 32B load.
__device__ __forceinline__ v16bf load_bfrag(const unsigned short* base, int ldk,
                                            int col0, int k0, int lane) {
  const int c    = col0 + (lane & 15);
  const int koff = k0 + ((lane >> 4) << 4);
  return *(const v16bf*)(base + (size_t)c * ldk + koff);
}
__device__ __forceinline__ v16bf load_bfrag_clamp(const unsigned short* base, int ldk,
                                                  int col0, int k0, int lane, int maxrow) {
  int c = col0 + (lane & 15);
  c = c < 0 ? 0 : (c >= maxrow ? maxrow - 1 : c);
  const int koff = k0 + ((lane >> 4) << 4);
  return *(const v16bf*)(base + (size_t)c * ldk + koff);
}

__device__ __forceinline__ float red_max16(float v) {
  v = fmaxf(v, __shfl_xor(v, 1));
  v = fmaxf(v, __shfl_xor(v, 2));
  v = fmaxf(v, __shfl_xor(v, 4));
  v = fmaxf(v, __shfl_xor(v, 8));
  return v;
}
__device__ __forceinline__ float red_sum16(float v) {
  v += __shfl_xor(v, 1);
  v += __shfl_xor(v, 2);
  v += __shfl_xor(v, 4);
  v += __shfl_xor(v, 8);
  return v;
}

// ---------------- conversion kernels ----------------
__global__ void cvt_f32_bf16_kernel(const float* __restrict__ src,
                                    unsigned short* __restrict__ dst, size_t n) {
  size_t i = (size_t)blockIdx.x * 256 + threadIdx.x;
  if (i < n) dst[i] = f2bf(src[i]);
}

// w: [K=1024][N=1024] f32 -> wt: [N][K] bf16   (weight transpose for B fragments)
__global__ void wtrans_kernel(const float* __restrict__ w,
                              unsigned short* __restrict__ wt) {
  size_t idx = (size_t)blockIdx.x * 256 + threadIdx.x;
  if (idx >= (size_t)DMODEL * DMODEL) return;
  int k = (int)(idx >> 10), j = (int)(idx & 1023);
  wt[(size_t)j * DMODEL + k] = f2bf(w[idx]);
}

// seg_mat [q,k,b,2] one-hot + attn_mask [q,k,b,1] -> 1 byte per (b,i,j):
// bit0 = different-segment selector, bit1 = masked.  12 B/elem -> 1 B/elem.
__global__ void pack_extra_kernel(const float* __restrict__ seg_mat,
                                  const float* __restrict__ attn_mask,
                                  unsigned char* __restrict__ packed) {
  size_t idx = (size_t)blockIdx.x * 256 + threadIdx.x;
  if (idx >= (size_t)BATCH * QLEN * KLENC) return;
  const int j = (int)(idx & (KLENC - 1));
  const int i = (int)((idx >> 11) & (QLEN - 1));
  const int b = (int)(idx >> 21);
  const size_t e = ((size_t)i * KLENC + j) * BATCH + b;
  const unsigned char sel = seg_mat[e * 2 + 1] != 0.0f ? 1 : 0;
  const unsigned char msk = attn_mask[e] != 0.0f ? 2 : 0;
  packed[idx] = (unsigned char)(sel | msk);
}

// ---------------- projection GEMM (WMMA bf16) ----------------
// A: [Mrows][K] bf16 row-major.  Bw: [1024][K] bf16 (pre-transposed weights).
// mode 0: store bf16 [bn][S][64]   (Q raw / K / K_r)
// mode 1: store bf16 [bn][64][S]   (V transposed for PV B-fragments)
// mode 2: store f32  [row][1024] + residual (output projection)
__global__ __launch_bounds__(256)
void proj_gemm_kernel(const unsigned short* __restrict__ A,
                      const unsigned short* __restrict__ Bw,
                      unsigned short* __restrict__ outb,
                      float* __restrict__ outf,
                      const float* __restrict__ resid,
                      int K, int S, int mode) {
  const int lane = threadIdx.x & 31;
  const int wave = threadIdx.x >> 5;
  const int m0 = blockIdx.x * 32 + (wave & 1) * 16;
  const int n0 = blockIdx.y * 64 + (wave >> 1) * 16;

  v8f acc = zero_v8f();
  for (int k0 = 0; k0 < K; k0 += 32) {
    v16bf a = load_afrag(A, K, m0, k0, lane);
    v16bf b = load_bfrag(Bw, K, n0, k0, lane);
    acc = wmma_bf16(a, b, acc);
  }
  const int half = lane >> 4, nn = lane & 15;
#pragma unroll
  for (int r = 0; r < 8; ++r) {
    const int m    = r + 8 * half;
    const int rowg = m0 + m;
    const int colg = n0 + nn;
    if (mode == 2) {
      outf[(size_t)rowg * DMODEL + colg] =
          acc[r] + resid[(size_t)rowg * DMODEL + colg];
    } else {
      const int b_ = rowg & 1, i = rowg >> 1;     // row = i*BATCH + b
      const int nh = colg >> 6, d = colg & 63;    // col = nh*64 + d
      const int bn = b_ * NHD + nh;
      if (mode == 0)
        outb[((size_t)bn * S + i) * DHD + d] = f2bf(acc[r]);
      else
        outb[((size_t)bn * DHD + d) * S + i] = f2bf(acc[r]);
    }
  }
}

// Q + r_w_bias -> Qc ;  Q + r_r_bias -> Qb   (bf16, layout [bn][QLEN][64])
__global__ void qbias_kernel(const unsigned short* __restrict__ Q,
                             const float* __restrict__ rwb,
                             const float* __restrict__ rrb,
                             unsigned short* __restrict__ Qc,
                             unsigned short* __restrict__ Qb) {
  size_t idx = (size_t)blockIdx.x * 256 + threadIdx.x;
  if (idx >= (size_t)BATCH * NHD * QLEN * DHD) return;
  const int d  = (int)(idx & 63);
  const int bn = (int)(idx / ((size_t)QLEN * DHD));
  const int n  = bn & (NHD - 1);
  const float q = bf2f(Q[idx]);
  Qc[idx] = f2bf(q + rwb[n * DHD + d]);
  Qb[idx] = f2bf(q + rrb[n * DHD + d]);
}

// EF[s][bn][i] = sum_d (Q[bn][i][d] + r_s_bias[n][d]) * seg_embed[s][n][d]
__global__ void ef_kernel(const unsigned short* __restrict__ Q,
                          const float* __restrict__ rsb,
                          const float* __restrict__ seg_embed,
                          float* __restrict__ EF) {
  int tid = blockIdx.x * 256 + threadIdx.x;          // 2*32*1024 threads
  if (tid >= 2 * BATCH * NHD * QLEN) return;
  const int i  = tid & (QLEN - 1);
  const int bn = (tid >> 10) & 31;
  const int s  = tid >> 15;
  const int n  = bn & (NHD - 1);
  float acc = 0.f;
#pragma unroll 8
  for (int d = 0; d < DHD; ++d)
    acc += (bf2f(Q[((size_t)bn * QLEN + i) * DHD + d]) + rsb[n * DHD + d]) *
           seg_embed[((size_t)s * NHD + n) * DHD + d];
  EF[(size_t)s * (BATCH * NHD * QLEN) + (size_t)bn * QLEN + i] = acc;
}

// ---------------- fused flash attention ----------------
// grid = (QLEN/128, BATCH*NHD); block = 256 (8 waves, one 16-row q-tile each)
__global__ __launch_bounds__(256)
void flash_attn_kernel(const unsigned short* __restrict__ Qc,
                       const unsigned short* __restrict__ Qb,
                       const unsigned short* __restrict__ Kb,
                       const unsigned short* __restrict__ VT,
                       const unsigned short* __restrict__ Rb,
                       const float* __restrict__ EF,
                       const unsigned char* __restrict__ packed,
                       unsigned short* __restrict__ vec) {
  __shared__ __align__(16) float          bandLds[8][16 * 32];  // per-wave diag band
  __shared__ __align__(32) unsigned short pLds[8][16 * 32];     // per-wave P tile

  const int lane = threadIdx.x & 31;
  const int wave = threadIdx.x >> 5;
  const int bn   = blockIdx.y;                 // bn = b*16 + n
  const int b    = bn >> 4;
  const int nh   = bn & 15;
  const int i0   = (blockIdx.x * 8 + wave) * 16;

  float*          band = bandLds[wave];
  unsigned short* pbuf = pLds[wave];

  const unsigned short* Qc_b = Qc + (size_t)bn * QLEN * DHD;
  const unsigned short* Qb_b = Qb + (size_t)bn * QLEN * DHD;
  const unsigned short* K_b  = Kb + (size_t)bn * KLENC * DHD;
  const unsigned short* VT_b = VT + (size_t)bn * DHD * KLENC;
  const unsigned short* R_b  = Rb + (size_t)bn * RLENC * DHD;
  const float* ef0 = EF + (size_t)bn * QLEN;
  const float* ef1 = EF + (size_t)(BATCH * NHD * QLEN) + (size_t)bn * QLEN;
  const unsigned char* pk_b = packed + (size_t)b * QLEN * KLENC;

  const int half = lane >> 4, nn = lane & 15;

  // persistent Q fragments (content + position biased), d-chunks {0,32}
  v16bf qc0 = load_afrag(Qc_b, DHD, i0, 0, lane);
  v16bf qc1 = load_afrag(Qc_b, DHD, i0, 32, lane);
  v16bf qb0 = load_afrag(Qb_b, DHD, i0, 0, lane);
  v16bf qb1 = load_afrag(Qb_b, DHD, i0, 32, lane);

  // per-row segment head-terms are constant across the whole KLEN sweep
  float e0r[8], e1r[8];
#pragma unroll
  for (int r = 0; r < 8; ++r) {
    const int gi = i0 + r + 8 * half;
    e0r[r] = ef0[gi];
    e1r[r] = ef1[gi];
  }

  v8f o0 = zero_v8f(), o1 = zero_v8f(), o2 = zero_v8f(), o3 = zero_v8f();
  float m_run[8], l_run[8];
#pragma unroll
  for (int r = 0; r < 8; ++r) { m_run[r] = -3.0e38f; l_run[r] = 0.f; }

  for (int j0 = 0; j0 < KLENC; j0 += 32) {
    if (j0 + 32 < KLENC) {                      // CDNA5 global_prefetch_b8 hints
      __builtin_prefetch(K_b + (size_t)(j0 + 32) * DHD, 0, 1);
      int pr = QLEN + (j0 + 32) - i0 - 15;
      if (pr < 0) pr = 0;
      if (pr > RLENC - 1) pr = RLENC - 1;
      __builtin_prefetch(R_b + (size_t)pr * DHD, 0, 1);
    }

    v8f sc[2];
#pragma unroll
    for (int sub = 0; sub < 2; ++sub) {
      const int js = j0 + sub * 16;
      // --- content term:  AC = Qc_tile @ K_tile^T ---
      v8f ac = zero_v8f();
      ac = wmma_bf16(qc0, load_bfrag(K_b, DHD, js, 0, lane), ac);
      ac = wmma_bf16(qc1, load_bfrag(K_b, DHD, js, 32, lane), ac);

      // --- position term: band T = Qb_tile @ R[pbase .. pbase+31]^T ---
      // bd[m][n] = T[m][15 + n - m]  (relative shift as anti-diagonal extract)
      const int pbase = QLEN + js - i0 - 15;
      v8f t0 = zero_v8f(), t1 = zero_v8f();
      t0 = wmma_bf16(qb0, load_bfrag_clamp(R_b, DHD, pbase,      0, lane, RLENC), t0);
      t0 = wmma_bf16(qb1, load_bfrag_clamp(R_b, DHD, pbase,     32, lane, RLENC), t0);
      t1 = wmma_bf16(qb0, load_bfrag_clamp(R_b, DHD, pbase + 16, 0, lane, RLENC), t1);
      t1 = wmma_bf16(qb1, load_bfrag_clamp(R_b, DHD, pbase + 16, 32, lane, RLENC), t1);

      // spill band to LDS (intra-wave; DS ops are in-order per wave)
#pragma unroll
      for (int r = 0; r < 8; ++r) {
        const int m = r + 8 * half;
        band[m * 32 + nn]      = t0[r];
        band[m * 32 + 16 + nn] = t1[r];
      }
      asm volatile("s_wait_dscnt 0" ::: "memory");

      // gather diagonal + packed segment/mask byte -> scaled scores
#pragma unroll
      for (int r = 0; r < 8; ++r) {
        const int m  = r + 8 * half;
        const int gi = i0 + m;
        const int gj = js + nn;
        const float bd = band[m * 32 + (15 + nn - m)];
        const unsigned char pv = pk_b[(size_t)gi * KLENC + gj];
        const float efv = (pv & 1) ? e1r[r] : e0r[r];
        const float mt  = (pv & 2) ? -1.0e30f : 0.0f;
        ac[r] = (ac[r] + bd + efv) * SCALEF + mt;
      }
      sc[sub] = ac;
    }

    // --- online softmax over the 32 new columns ---
    float alpha[8];
#pragma unroll
    for (int r = 0; r < 8; ++r) {
      float mx   = red_max16(fmaxf(sc[0][r], sc[1][r]));
      float mnew = fmaxf(m_run[r], mx);
      alpha[r]   = __expf(m_run[r] - mnew);
      float p0   = __expf(sc[0][r] - mnew);
      float p1   = __expf(sc[1][r] - mnew);
      l_run[r]   = l_run[r] * alpha[r] + red_sum16(p0 + p1);
      m_run[r]   = mnew;
      const int m = r + 8 * half;
      pbuf[m * 32 + nn]      = f2bf(p0);
      pbuf[m * 32 + 16 + nn] = f2bf(p1);
    }
#pragma unroll
    for (int r = 0; r < 8; ++r) {
      o0[r] = o0[r] * alpha[r];
      o1[r] = o1[r] * alpha[r];
      o2[r] = o2[r] * alpha[r];
      o3[r] = o3[r] * alpha[r];
    }
    asm volatile("s_wait_dscnt 0" ::: "memory");

    // P (C-layout) -> A-fragment relayout via LDS, then PV WMMAs
    v16bf pa = load_afrag(pbuf, 32, 0, 0, lane);
    o0 = wmma_bf16(pa, load_bfrag(VT_b, KLENC, 0,  j0, lane), o0);
    o1 = wmma_bf16(pa, load_bfrag(VT_b, KLENC, 16, j0, lane), o1);
    o2 = wmma_bf16(pa, load_bfrag(VT_b, KLENC, 32, j0, lane), o2);
    o3 = wmma_bf16(pa, load_bfrag(VT_b, KLENC, 48, j0, lane), o3);
  }

  // finalize: attn_vec row-normalized, bf16, layout [i*BATCH+b][nh*64+d]
#pragma unroll
  for (int r = 0; r < 8; ++r) {
    const int m   = r + 8 * half;
    const int gi  = i0 + m;
    const float s = 1.0f / l_run[r];
    unsigned short* vr =
        vec + ((size_t)(gi * BATCH + b)) * DMODEL + nh * DHD + nn;
    vr[0]  = f2bf(o0[r] * s);
    vr[16] = f2bf(o1[r] * s);
    vr[32] = f2bf(o2[r] * s);
    vr[48] = f2bf(o3[r] * s);
  }
}

// ---------------- LayerNorm ----------------
__global__ __launch_bounds__(256)
void ln_kernel(const float* __restrict__ x, const float* __restrict__ gamma,
               const float* __restrict__ beta, float* __restrict__ out) {
  __shared__ float s1[256], s2[256];
  const int row = blockIdx.x;
  const float* xr = x + (size_t)row * DMODEL;
  float a = 0.f, b = 0.f;
  for (int c = threadIdx.x; c < DMODEL; c += 256) {
    float v = xr[c];
    a += v; b += v * v;
  }
  s1[threadIdx.x] = a; s2[threadIdx.x] = b;
  __syncthreads();
  for (int off = 128; off > 0; off >>= 1) {
    if ((int)threadIdx.x < off) {
      s1[threadIdx.x] += s1[threadIdx.x + off];
      s2[threadIdx.x] += s2[threadIdx.x + off];
    }
    __syncthreads();
  }
  const float mu  = s1[0] * (1.0f / DMODEL);
  const float var = s2[0] * (1.0f / DMODEL) - mu * mu;
  const float rs  = rsqrtf(var + 1e-12f);
  for (int c = threadIdx.x; c < DMODEL; c += 256)
    out[(size_t)row * DMODEL + c] = (xr[c] - mu) * rs * gamma[c] + beta[c];
}

// ---------------- host launch ----------------
extern "C" void kernel_launch(void* const* d_in, const int* in_sizes, int n_in,
                              void* d_out, int out_size, void* d_ws, size_t ws_size,
                              hipStream_t stream) {
  (void)in_sizes; (void)n_in; (void)out_size; (void)ws_size;

  const float* h        = (const float*)d_in[0];
  const float* mems     = (const float*)d_in[1];
  const float* r        = (const float*)d_in[2];
  const float* seg_mat  = (const float*)d_in[3];
  const float* attn_msk = (const float*)d_in[4];
  const float* q_w      = (const float*)d_in[5];
  const float* k_w      = (const float*)d_in[6];
  const float* v_w      = (const float*)d_in[7];
  const float* o_w      = (const float*)d_in[8];
  const float* r_w      = (const float*)d_in[9];
  const float* r_w_bias = (const float*)d_in[10];
  const float* r_r_bias = (const float*)d_in[11];
  const float* r_s_bias = (const float*)d_in[12];
  const float* seg_emb  = (const float*)d_in[13];
  const float* ln_gamma = (const float*)d_in[14];
  const float* ln_beta  = (const float*)d_in[15];
  float* out = (float*)d_out;

  char* wsb = (char*)d_ws;
  size_t off = 0;
  auto alloc = [&](size_t bytes) -> void* {
    void* p = wsb + off;
    off = (off + bytes + 255) & ~(size_t)255;
    return p;
  };

  const size_t HB_N   = (size_t)QLEN * BATCH * DMODEL;   // 2 M
  const size_t CAT_N  = (size_t)KLENC * BATCH * DMODEL;  // 4 M
  const size_t RB_N   = (size_t)RLENC * BATCH * DMODEL;  // 6 M
  const size_t W_N    = (size_t)DMODEL * DMODEL;         // 1 M
  const size_t QH_N   = (size_t)BATCH * NHD * QLEN * DHD;   // 2 M
  const size_t KH_N   = (size_t)BATCH * NHD * KLENC * DHD;  // 4 M
  const size_t RH_N   = (size_t)BATCH * NHD * RLENC * DHD;  // 6 M
  const size_t PK_N   = (size_t)BATCH * QLEN * KLENC;       // 4 M bytes

  unsigned short* hb   = (unsigned short*)alloc(HB_N * 2);
  unsigned short* catb = (unsigned short*)alloc(CAT_N * 2);
  unsigned short* rb   = (unsigned short*)alloc(RB_N * 2);
  unsigned short* wtq  = (unsigned short*)alloc(W_N * 2);
  unsigned short* wtk  = (unsigned short*)alloc(W_N * 2);
  unsigned short* wtv  = (unsigned short*)alloc(W_N * 2);
  unsigned short* wtr  = (unsigned short*)alloc(W_N * 2);
  unsigned short* wo   = (unsigned short*)alloc(W_N * 2);
  unsigned short* qraw = (unsigned short*)alloc(QH_N * 2);
  unsigned short* qcb  = (unsigned short*)alloc(QH_N * 2);
  unsigned short* qbb  = (unsigned short*)alloc(QH_N * 2);
  unsigned short* kb   = (unsigned short*)alloc(KH_N * 2);
  unsigned short* vtb  = (unsigned short*)alloc(KH_N * 2);
  unsigned short* rpb  = (unsigned short*)alloc(RH_N * 2);
  float*          efb  = (float*)alloc((size_t)2 * BATCH * NHD * QLEN * 4);
  unsigned short* vecb = (unsigned short*)alloc(HB_N * 2);
  float*          attf = (float*)alloc(HB_N * 4);
  unsigned char*  pkb  = (unsigned char*)alloc(PK_N);

  const dim3 blk(256);
  auto nb = [](size_t n) { return dim3((unsigned)((n + 255) / 256)); };

  // 1) bf16 staging: h, cat=[mems;h], r ; pack seg/mask bits
  cvt_f32_bf16_kernel<<<nb(HB_N), blk, 0, stream>>>(h, hb, HB_N);
  cvt_f32_bf16_kernel<<<nb(HB_N), blk, 0, stream>>>(mems, catb, HB_N);
  cvt_f32_bf16_kernel<<<nb(HB_N), blk, 0, stream>>>(h, catb + HB_N, HB_N);
  cvt_f32_bf16_kernel<<<nb(RB_N), blk, 0, stream>>>(r, rb, RB_N);
  pack_extra_kernel<<<nb(PK_N), blk, 0, stream>>>(seg_mat, attn_msk, pkb);

  // 2) weights: transpose to [N][K] bf16 (B-fragment friendly); o_w is already [N][K]
  wtrans_kernel<<<nb(W_N), blk, 0, stream>>>(q_w, wtq);
  wtrans_kernel<<<nb(W_N), blk, 0, stream>>>(k_w, wtk);
  wtrans_kernel<<<nb(W_N), blk, 0, stream>>>(v_w, wtv);
  wtrans_kernel<<<nb(W_N), blk, 0, stream>>>(r_w, wtr);
  cvt_f32_bf16_kernel<<<nb(W_N), blk, 0, stream>>>(o_w, wo, W_N);

  // 3) projection GEMMs (WMMA bf16)
  proj_gemm_kernel<<<dim3(2048 / 32, 16), blk, 0, stream>>>(
      hb, wtq, qraw, nullptr, nullptr, DMODEL, QLEN, 0);
  proj_gemm_kernel<<<dim3(4096 / 32, 16), blk, 0, stream>>>(
      catb, wtk, kb, nullptr, nullptr, DMODEL, KLENC, 0);
  proj_gemm_kernel<<<dim3(4096 / 32, 16), blk, 0, stream>>>(
      catb, wtv, vtb, nullptr, nullptr, DMODEL, KLENC, 1);
  proj_gemm_kernel<<<dim3(6144 / 32, 16), blk, 0, stream>>>(
      rb, wtr, rpb, nullptr, nullptr, DMODEL, RLENC, 0);

  // 4) biased Q copies + segment head-term EF
  qbias_kernel<<<nb(QH_N), blk, 0, stream>>>(qraw, r_w_bias, r_r_bias, qcb, qbb);
  ef_kernel<<<nb((size_t)2 * BATCH * NHD * QLEN), blk, 0, stream>>>(
      qraw, r_s_bias, seg_emb, efb);

  // 5) fused flash attention (AC + rel-shift BD band + EF + mask + online softmax + PV)
  flash_attn_kernel<<<dim3(QLEN / 128, BATCH * NHD), blk, 0, stream>>>(
      qcb, qbb, kb, vtb, rpb, efb, pkb, vecb);

  // 6) output projection + residual (WMMA bf16, f32 out)
  proj_gemm_kernel<<<dim3(2048 / 32, 16), blk, 0, stream>>>(
      vecb, wo, nullptr, attf, h, DMODEL, QLEN, 2);

  // 7) LayerNorm
  ln_kernel<<<dim3(QLEN * BATCH), blk, 0, stream>>>(attf, ln_gamma, ln_beta, out);
}
